// MultiMPN_5111011082634
// MI455X (gfx1250) — compile-verified, matrix operand order checked by reading, not run on hardware
//
#include <hip/hip_runtime.h>
#include <hip/hip_bf16.h>
#include <stdint.h>

// ---------------------------------------------------------------------------
// MultiMPN on MI455X (gfx1250): bf16 WMMA edge/node GEMMs + atomic segment-sum
// Sizes are static and tile-exact (N=20000, 2E=640000, both % 16 == 0), so all
// per-element bounds guards are removed -> no exec-mask dances around vmem.
// ---------------------------------------------------------------------------

typedef __attribute__((ext_vector_type(16))) __bf16    v16bf;
typedef __attribute__((ext_vector_type(8)))  float     v8f;
typedef __attribute__((ext_vector_type(4)))  unsigned  u32x4;

union Frag16 { v16bf v; u32x4 q[2]; };

__device__ __forceinline__ unsigned short f2bf(float f) {
  unsigned u = __builtin_bit_cast(unsigned, f);
  u += 0x7FFFu + ((u >> 16) & 1u);           // round-to-nearest-even
  return (unsigned short)(u >> 16);
}

// Wave-scope LDS ordering: tiles are wave-private, so a full __syncthreads is
// unnecessary; use the CDNA5 split DS counter directly.
__device__ __forceinline__ void wave_lds_fence() {
  __builtin_amdgcn_wave_barrier();
  asm volatile("s_wait_dscnt 0" ::: "memory");
  __builtin_amdgcn_wave_barrier();
}

// Directed edge e in [0,2E): src = ei[e]; dst = ei[E+e] (e<E) else ei[e-E].
__device__ __forceinline__ int edge_dst(const int* ei, int e, int nE) {
  return (e < nE) ? ei[nE + e] : ei[e - nE];
}

// ---------------------------------------------------------------------------
// Edge MLP: per 16-edge tile, A = [x[dst] | x[src] | eattr] (16 x KPAD bf16),
// GEMM1 (KPAD x 64) -> bias+ReLU -> GEMM2 (64 x HOUTP) -> bias -> atomic
// scatter-add into out[dst].  nE2 must be a multiple of 16 (it is: 640000).
// ---------------------------------------------------------------------------
template<int HIN, int HOUT>
__global__ __launch_bounds__(128) void edge_mlp_kernel(
    const unsigned short* __restrict__ xb,   // N x HIN   (bf16)
    const int*            __restrict__ ei,   // flat 2 x E
    const float*          __restrict__ ea,   // E x 5     (f32)
    const unsigned short* __restrict__ W1T,  // 64 x KPAD (bf16, N-major)
    const float*          __restrict__ b1,   // 64
    const unsigned short* __restrict__ W2T,  // HOUTP x 64 (bf16, N-major)
    const float*          __restrict__ b2,   // HOUT
    float*                __restrict__ out,  // N x HOUT accumulator (pre-zeroed)
    int nE2, int nE)
{
  constexpr int KDIM  = 2 * HIN + 5;
  constexpr int KPAD  = (KDIM + 31) & ~31;
  constexpr int NCH   = KPAD / 32;
  constexpr int HOUTP = (HOUT + 15) & ~15;
  constexpr int NT2   = HOUTP / 16;
  constexpr int NWAVE = 4;

  __shared__ alignas(16) unsigned short Atile[NWAVE][16 * KPAD];
  __shared__ alignas(16) unsigned short Htile[NWAVE][16 * 64];
  __shared__ int sDst[NWAVE][16];

  const int wave = threadIdx.x >> 5;
  const int lane = threadIdx.x & 31;
  const int half = lane >> 4;
  const int l16  = lane & 15;

  // Loop-invariant bias values (hoisted: loop may be taken multiple times).
  float bias1[4];
  #pragma unroll
  for (int n = 0; n < 4; ++n) bias1[n] = b1[n * 16 + l16];
  float bias2[NT2];
  #pragma unroll
  for (int n = 0; n < NT2; ++n) {
    int col = n * 16 + l16;
    bias2[n] = (col < HOUT) ? b2[col] : 0.0f;
  }

  const int nTiles = nE2 >> 4;            // exact
  for (int tile = blockIdx.x * NWAVE + wave; tile < nTiles;
       tile += (int)gridDim.x * NWAVE) {

    int eLane = tile * 16 + l16;
    int srcL  = ei[eLane];
    int dstL  = edge_dst(ei, eLane, nE);
    if (half == 0) sDst[wave][l16] = dstL;
    // src also staged so the gather loop can read any row's indices
    __shared__ int sSrc[NWAVE][16];
    if (half == 0) sSrc[wave][l16] = srcL;
    wave_lds_fence();

    // Gather concat rows into the A tile (bf16). Full tile, no guards.
    #pragma unroll 4
    for (int idx = lane; idx < 16 * KPAD; idx += 32) {
      int m = idx / KPAD, k = idx - m * KPAD;
      int e = tile * 16 + m;
      unsigned short v = 0;
      if      (k < HIN)     v = xb[sDst[wave][m] * HIN + k];
      else if (k < 2 * HIN) v = xb[sSrc[wave][m] * HIN + (k - HIN)];
      else if (k < KDIM)    v = f2bf(ea[(e < nE ? e : e - nE) * 5 + (k - 2 * HIN)]);
      Atile[wave][idx] = v;
    }
    wave_lds_fence();

    // GEMM1: hidden (16 x 64), 4 column tiles.
    v8f acc1[4] = {};
    #pragma unroll
    for (int c = 0; c < NCH; ++c) {
      Frag16 a;
      const unsigned short* ap = &Atile[wave][l16 * KPAD + c * 32 + half * 8];
      a.q[0] = *reinterpret_cast<const u32x4*>(ap);
      a.q[1] = *reinterpret_cast<const u32x4*>(ap + 16);
      #pragma unroll
      for (int n = 0; n < 4; ++n) {
        Frag16 b;
        const unsigned short* bp = &W1T[(n * 16 + l16) * KPAD + c * 32 + half * 8];
        b.q[0] = *reinterpret_cast<const u32x4*>(bp);
        b.q[1] = *reinterpret_cast<const u32x4*>(bp + 16);
        acc1[n] = __builtin_amdgcn_wmma_f32_16x16x32_bf16(
            false, a.v, false, b.v, (short)0, acc1[n], false, false);
      }
    }

    // bias + ReLU -> Htile (bf16) for the second GEMM's A fragments.
    #pragma unroll
    for (int n = 0; n < 4; ++n) {
      int col = n * 16 + l16;
      #pragma unroll
      for (int r = 0; r < 8; ++r) {
        int row = r + half * 8;
        float h = fmaxf(acc1[n][r] + bias1[n], 0.0f);
        Htile[wave][row * 64 + col] = f2bf(h);
      }
    }
    wave_lds_fence();

    // GEMM2: (16 x 64) x (64 x HOUTP).
    v8f acc2[NT2] = {};
    #pragma unroll
    for (int c = 0; c < 2; ++c) {
      Frag16 a;
      const unsigned short* ap = &Htile[wave][l16 * 64 + c * 32 + half * 8];
      a.q[0] = *reinterpret_cast<const u32x4*>(ap);
      a.q[1] = *reinterpret_cast<const u32x4*>(ap + 16);
      #pragma unroll
      for (int n = 0; n < NT2; ++n) {
        Frag16 b;
        const unsigned short* bp = &W2T[(n * 16 + l16) * 64 + c * 32 + half * 8];
        b.q[0] = *reinterpret_cast<const u32x4*>(bp);
        b.q[1] = *reinterpret_cast<const u32x4*>(bp + 16);
        acc2[n] = __builtin_amdgcn_wmma_f32_16x16x32_bf16(
            false, a.v, false, b.v, (short)0, acc2[n], false, false);
      }
    }

    // Per-edge message (+b2) scatter-added into out[dst] (segment_sum).
    #pragma unroll
    for (int n = 0; n < NT2; ++n) {
      int col = n * 16 + l16;
      if (HOUT % 16 == 0 || col < HOUT) {
        #pragma unroll
        for (int r = 0; r < 8; ++r) {
          int row = r + half * 8;
          atomicAdd(&out[sDst[wave][row] * HOUT + col], acc2[n][r] + bias2[n]);
        }
      }
    }
    wave_lds_fence();   // protect sDst/sSrc/Atile before next iteration
  }
}

// ---------------------------------------------------------------------------
// Node GEMM: out[16-node tile] (+)= xb @ W (64x64). A fragments loaded
// straight from global (rows are 128B contiguous); W pre-transposed N-major.
// nNodes must be a multiple of 16 (it is: 20000).
// ---------------------------------------------------------------------------
template<bool ACC>
__global__ __launch_bounds__(128) void node_gemm_kernel(
    const unsigned short* __restrict__ xb,  // nNodes x 64 (bf16)
    const unsigned short* __restrict__ WT,  // 64 x 64 (bf16, N-major)
    float*                __restrict__ out, // nNodes x 64
    int nNodes)
{
  const int wave = threadIdx.x >> 5;
  const int lane = threadIdx.x & 31;
  const int half = lane >> 4;
  const int l16  = lane & 15;
  const int nTiles = nNodes >> 4;          // exact

  // B fragments are tile-invariant: load the whole 64x64 weight once.
  Frag16 bfr[2][4];
  #pragma unroll
  for (int c = 0; c < 2; ++c)
    #pragma unroll
    for (int n = 0; n < 4; ++n) {
      const unsigned short* bp = &WT[(n * 16 + l16) * 64 + c * 32 + half * 8];
      bfr[c][n].q[0] = *reinterpret_cast<const u32x4*>(bp);
      bfr[c][n].q[1] = *reinterpret_cast<const u32x4*>(bp + 16);
    }

  for (int tile = blockIdx.x * 4 + wave; tile < nTiles;
       tile += (int)gridDim.x * 4) {
    v8f acc[4] = {};
    if (ACC) {
      #pragma unroll
      for (int n = 0; n < 4; ++n)
        #pragma unroll
        for (int r = 0; r < 8; ++r) {
          int node = tile * 16 + r + half * 8;
          acc[n][r] = out[node * 64 + n * 16 + l16];
        }
    }
    int nodeA = tile * 16 + l16;
    #pragma unroll
    for (int c = 0; c < 2; ++c) {
      Frag16 a;
      const unsigned short* ap = &xb[nodeA * 64 + c * 32 + half * 8];
      a.q[0] = *reinterpret_cast<const u32x4*>(ap);
      a.q[1] = *reinterpret_cast<const u32x4*>(ap + 16);
      #pragma unroll
      for (int n = 0; n < 4; ++n) {
        acc[n] = __builtin_amdgcn_wmma_f32_16x16x32_bf16(
            false, a.v, false, bfr[c][n].v, (short)0, acc[n], false, false);
      }
    }
    #pragma unroll
    for (int n = 0; n < 4; ++n)
      #pragma unroll
      for (int r = 0; r < 8; ++r) {
        int node = tile * 16 + r + half * 8;
        out[node * 64 + n * 16 + l16] = acc[n][r];
      }
  }
}

// ---------------------------------------------------------------------------
// Small helper kernels
// ---------------------------------------------------------------------------
__global__ void zero_kernel(float* __restrict__ p, int n) {
  int i = blockIdx.x * blockDim.x + threadIdx.x;
  if (i < n) p[i] = 0.0f;
}

__global__ void deg_kernel(const int* __restrict__ ei, float* __restrict__ deg,
                           int nE2, int nE) {
  int e = blockIdx.x * blockDim.x + threadIdx.x;
  if (e < nE2) atomicAdd(&deg[edge_dst(ei, e, nE)], 1.0f);
}

__global__ void dis_kernel(const float* __restrict__ deg, float* __restrict__ dis,
                           int n) {
  int i = blockIdx.x * blockDim.x + threadIdx.x;
  if (i < n) dis[i] = (deg[i] > 0.0f) ? rsqrtf(deg[i]) : 0.0f;
}

__global__ void norm_kernel(const int* __restrict__ ei, const float* __restrict__ dis,
                            float* __restrict__ nrm, int nE2, int nE) {
  int e = blockIdx.x * blockDim.x + threadIdx.x;
  if (e < nE2) nrm[e] = dis[ei[e]] * dis[edge_dst(ei, e, nE)];
}

// x0 = data_x[:, 4:10] as bf16 (N x 6)
__global__ void x0_kernel(const float* __restrict__ dx,
                          unsigned short* __restrict__ x0b, int n) {
  int idx = blockIdx.x * blockDim.x + threadIdx.x;
  if (idx < n * 6) {
    int i = idx / 6, j = idx - i * 6;
    x0b[idx] = f2bf(dx[i * 16 + 4 + j]);
  }
}

// WT[n*KPAD + k] = bf16(W[k*Ncols + n]), zero-padded to (NP x KPAD)
__global__ void wt_kernel(const float* __restrict__ W, unsigned short* __restrict__ WT,
                          int Kdim, int KPAD, int Ncols, int NP) {
  int idx = blockIdx.x * blockDim.x + threadIdx.x;
  if (idx < NP * KPAD) {
    int nn = idx / KPAD, k = idx - nn * KPAD;
    float v = (k < Kdim && nn < Ncols) ? W[k * Ncols + nn] : 0.0f;
    WT[idx] = f2bf(v);
  }
}

// 12 batched 64x64 transposes for tag_W
__global__ void tagwt_kernel(const float* __restrict__ W,
                             unsigned short* __restrict__ WT) {
  int idx = blockIdx.x * blockDim.x + threadIdx.x;
  if (idx < 12 * 4096) {
    int b = idx >> 12, r = idx & 4095;
    int nn = r >> 6, k = r & 63;
    WT[idx] = f2bf(W[b * 4096 + k * 64 + nn]);
  }
}

// hn[dst, c] += norm[e] * h[src, c]   (TAGConv propagation)
__global__ void prop_kernel(const float* __restrict__ h, const int* __restrict__ ei,
                            const float* __restrict__ nrm, float* __restrict__ hn,
                            int nE2, int nE) {
  int tid = blockIdx.x * blockDim.x + threadIdx.x;
  if (tid < nE2 * 64) {
    int e = tid >> 6, c = tid & 63;
    int s = ei[e];
    int d = edge_dst(ei, e, nE);
    atomicAdd(&hn[d * 64 + c], nrm[e] * h[s * 64 + c]);
  }
}

__global__ void tobf_kernel(const float* __restrict__ h,
                            unsigned short* __restrict__ hb, int n) {
  int i = blockIdx.x * blockDim.x + threadIdx.x;
  if (i < n) hb[i] = f2bf(h[i]);
}

// x = relu(in (+ bias per col)); emit both f32 and bf16 copies
__global__ void finalize_kernel(const float* __restrict__ in,
                                const float* __restrict__ bias,
                                float* __restrict__ of,
                                unsigned short* __restrict__ ob, int rows) {
  int idx = blockIdx.x * blockDim.x + threadIdx.x;
  if (idx < rows * 64) {
    int c = idx & 63;
    float v = in[idx] + (bias ? bias[c] : 0.0f);
    v = fmaxf(v, 0.0f);
    of[idx] = v;
    ob[idx] = f2bf(v);
  }
}

// ---------------------------------------------------------------------------
// Host-side orchestration
// ---------------------------------------------------------------------------
extern "C" void kernel_launch(void* const* d_in, const int* in_sizes, int n_in,
                              void* d_out, int out_size, void* d_ws, size_t ws_size,
                              hipStream_t stream) {
  (void)in_sizes; (void)n_in; (void)out_size; (void)ws_size;

  const float* data_x = (const float*)d_in[0];
  const int*   ei     = (const int*)d_in[1];
  const float* eattr  = (const float*)d_in[2];
  const float* ea0W1  = (const float*)d_in[3];
  const float* ea0b1  = (const float*)d_in[4];
  const float* ea0W2  = (const float*)d_in[5];
  const float* ea0b2  = (const float*)d_in[6];
  const float* eamW1  = (const float*)d_in[7];   // 2 x 133 x 64
  const float* eamb1  = (const float*)d_in[8];   // 2 x 64
  const float* eamW2  = (const float*)d_in[9];   // 2 x 64 x 64
  const float* eamb2  = (const float*)d_in[10];  // 2 x 64
  const float* ealW1  = (const float*)d_in[11];  // 133 x 64
  const float* ealb1  = (const float*)d_in[12];
  const float* ealW2  = (const float*)d_in[13];  // 64 x 6
  const float* ealb2  = (const float*)d_in[14];
  const float* tagW   = (const float*)d_in[15];  // 3 x 4 x 64 x 64
  const float* tagb   = (const float*)d_in[16];  // 3 x 64
  float* out = (float*)d_out;

  const int N = 20000, E = 320000, E2 = 640000;

  char* ws = (char*)d_ws;
  size_t off = 0;
  auto alloc = [&](size_t bytes) -> void* {
    void* p = ws + off;
    off = (off + bytes + 255) & ~(size_t)255;
    return p;
  };
  float* deg  = (float*)alloc((size_t)N * 4);
  float* dis  = (float*)alloc((size_t)N * 4);
  float* nrm  = (float*)alloc((size_t)E2 * 4);
  unsigned short* x0b = (unsigned short*)alloc((size_t)N * 6 * 2);
  unsigned short* xb  = (unsigned short*)alloc((size_t)N * 64 * 2);
  float* xf   = (float*)alloc((size_t)N * 64 * 4);
  float* accb = (float*)alloc((size_t)N * 64 * 4);
  float* h1   = (float*)alloc((size_t)N * 64 * 4);
  float* h2   = (float*)alloc((size_t)N * 64 * 4);
  unsigned short* hb    = (unsigned short*)alloc((size_t)N * 64 * 2);
  unsigned short* w1t0  = (unsigned short*)alloc(64 * 32 * 2);
  unsigned short* w2t0  = (unsigned short*)alloc(64 * 64 * 2);
  unsigned short* w1tm0 = (unsigned short*)alloc(64 * 160 * 2);
  unsigned short* w1tm1 = (unsigned short*)alloc(64 * 160 * 2);
  unsigned short* w2tm0 = (unsigned short*)alloc(64 * 64 * 2);
  unsigned short* w2tm1 = (unsigned short*)alloc(64 * 64 * 2);
  unsigned short* w1tl  = (unsigned short*)alloc(64 * 160 * 2);
  unsigned short* w2tl  = (unsigned short*)alloc(16 * 64 * 2);
  unsigned short* tagwt = (unsigned short*)alloc(12 * 4096 * 2);

  auto g1 = [](int n) { return dim3((unsigned)((n + 255) / 256)); };

  // ---- weight conversion + transpose (bf16, N-major) ----
  wt_kernel<<<g1(64 * 32), 256, 0, stream>>>(ea0W1, w1t0, 17, 32, 64, 64);
  wt_kernel<<<g1(64 * 64), 256, 0, stream>>>(ea0W2, w2t0, 64, 64, 64, 64);
  wt_kernel<<<g1(64 * 160), 256, 0, stream>>>(eamW1, w1tm0, 133, 160, 64, 64);
  wt_kernel<<<g1(64 * 160), 256, 0, stream>>>(eamW1 + 133 * 64, w1tm1, 133, 160, 64, 64);
  wt_kernel<<<g1(64 * 64), 256, 0, stream>>>(eamW2, w2tm0, 64, 64, 64, 64);
  wt_kernel<<<g1(64 * 64), 256, 0, stream>>>(eamW2 + 4096, w2tm1, 64, 64, 64, 64);
  wt_kernel<<<g1(64 * 160), 256, 0, stream>>>(ealW1, w1tl, 133, 160, 64, 64);
  wt_kernel<<<g1(16 * 64), 256, 0, stream>>>(ealW2, w2tl, 64, 64, 6, 16);
  tagwt_kernel<<<g1(12 * 4096), 256, 0, stream>>>(tagW, tagwt);

  // ---- graph normalization ----
  zero_kernel<<<g1(N), 256, 0, stream>>>(deg, N);
  deg_kernel<<<g1(E2), 256, 0, stream>>>(ei, deg, E2, E);
  dis_kernel<<<g1(N), 256, 0, stream>>>(deg, dis, N);
  norm_kernel<<<g1(E2), 256, 0, stream>>>(ei, dis, nrm, E2, E);
  x0_kernel<<<g1(N * 6), 256, 0, stream>>>(data_x, x0b, N);

  const int mlpBlocks  = (E2 / 16 + 3) / 4;  // 10000 blocks x 4 waves
  const int nodeBlocks = (N / 16 + 3) / 4;   // 313 blocks

  auto tag_layer = [&](int l) {
    const unsigned short* wt = tagwt + (size_t)l * 4 * 4096;
    node_gemm_kernel<false><<<nodeBlocks, 128, 0, stream>>>(xb, wt, accb, N);
    const float* hc = xf;
    float* hn = h1;
    for (int k = 1; k <= 3; ++k) {
      zero_kernel<<<g1(N * 64), 256, 0, stream>>>(hn, N * 64);
      prop_kernel<<<g1(E2 * 64), 256, 0, stream>>>(hc, ei, nrm, hn, E2, E);
      tobf_kernel<<<g1(N * 64), 256, 0, stream>>>(hn, hb, N * 64);
      node_gemm_kernel<true><<<nodeBlocks, 128, 0, stream>>>(hb, wt + k * 4096, accb, N);
      hc = hn;
      hn = (hn == h1) ? h2 : h1;
    }
    finalize_kernel<<<g1(N * 64), 256, 0, stream>>>(accb, tagb + l * 64, xf, xb, N);
  };

  // ---- stage 0: edge MLP (6 -> 64) + TAGConv 0 ----
  zero_kernel<<<g1(N * 64), 256, 0, stream>>>(accb, N * 64);
  edge_mlp_kernel<6, 64><<<mlpBlocks, 128, 0, stream>>>(
      x0b, ei, eattr, w1t0, ea0b1, w2t0, ea0b2, accb, E2, E);
  finalize_kernel<<<g1(N * 64), 256, 0, stream>>>(accb, nullptr, xf, xb, N);
  tag_layer(0);

  // ---- mid 0 ----
  zero_kernel<<<g1(N * 64), 256, 0, stream>>>(accb, N * 64);
  edge_mlp_kernel<64, 64><<<mlpBlocks, 128, 0, stream>>>(
      xb, ei, eattr, w1tm0, eamb1, w2tm0, eamb2, accb, E2, E);
  finalize_kernel<<<g1(N * 64), 256, 0, stream>>>(accb, nullptr, xf, xb, N);
  tag_layer(1);

  // ---- mid 1 ----
  zero_kernel<<<g1(N * 64), 256, 0, stream>>>(accb, N * 64);
  edge_mlp_kernel<64, 64><<<mlpBlocks, 128, 0, stream>>>(
      xb, ei, eattr, w1tm1, eamb1 + 64, w2tm1, eamb2 + 64, accb, E2, E);
  finalize_kernel<<<g1(N * 64), 256, 0, stream>>>(accb, nullptr, xf, xb, N);
  tag_layer(2);

  // ---- final edge MLP (64 -> 6), no ReLU, straight into d_out ----
  zero_kernel<<<g1(N * 6), 256, 0, stream>>>(out, N * 6);
  edge_mlp_kernel<64, 6><<<mlpBlocks, 128, 0, stream>>>(
      xb, ei, eattr, w1tl, ealb1, w2tl, ealb2, out, E2, E);
}
